// NCNC_6545530159542
// MI455X (gfx1250) — matrix operand order repaired and checked by reading, not compiled, not run on hardware
//
#include <hip/hip_runtime.h>
#include <math.h>

// ---------------- problem constants ----------------
#define NN      2048      // nodes
#define DD      128       // embed dim
#define BB      16        // link batch
#define K1      256       // MLP input features (2*D)
#define HH      512       // hidden
#define MAXROWS 4096      // cap on exclusive-neighbor rows (expected ~1300)
#define MAXC    4096      // cap on common-neighbor pairs (expected ~13)

typedef __bf16 v16bf __attribute__((ext_vector_type(16)));
typedef float  v8f   __attribute__((ext_vector_type(8)));

union FragBF { unsigned int u[8]; v16bf v; };

__device__ __forceinline__ unsigned short f2bf(float f) {
    unsigned int u = __builtin_bit_cast(unsigned int, f);
    u = (u + 0x7FFFu + ((u >> 16) & 1u)) >> 16;
    return (unsigned short)u;
}

__device__ __forceinline__ int eff_rows(const int* cnt, int fixedM) {
    int M = cnt ? *cnt : fixedM;
    return M > MAXROWS ? MAXROWS : M;
}

// ---------------- init / prep kernels ----------------
__global__ void zero_u32(unsigned int* p, int n) {
    int i = blockIdx.x * blockDim.x + threadIdx.x;
    if (i < n) p[i] = 0u;
}

// pack bool adjacency rows into 32-bit masks: mask[node*64 + wd]
__global__ void pack_adj(const unsigned char* __restrict__ adj, unsigned int* __restrict__ mask) {
    int idx = blockIdx.x * blockDim.x + threadIdx.x;
    if (idx >= NN * 64) return;
    int node = idx >> 6, wd = idx & 63;
    const unsigned char* p = adj + (size_t)node * NN + wd * 32;
    unsigned int w = 0;
    #pragma unroll
    for (int t = 0; t < 32; ++t) w |= (p[t] ? 1u : 0u) << t;
    mask[idx] = w;
}

// W [K,Nout] f32 row-major  ->  Wt [Nout,K] bf16 (transposed, for contiguous B-frag loads)
__global__ void prep_wt(const float* __restrict__ W, unsigned short* __restrict__ Wt,
                        int K, int Nout) {
    int idx = blockIdx.x * blockDim.x + threadIdx.x;
    if (idx >= K * Nout) return;
    int k = idx / Nout, n = idx % Nout;
    Wt[(size_t)n * K + k] = f2bf(W[idx]);
}

// ---------------- deterministic candidate compaction (single wave32) ----------------
__global__ void compact(const int* __restrict__ TL, const unsigned char* __restrict__ adj,
                        int* __restrict__ rb, int* __restrict__ ru, int* __restrict__ rref,
                        int* __restrict__ cb, int* __restrict__ cu, int* __restrict__ counts) {
    int lane = threadIdx.x;
    unsigned long long lt = (1ull << lane) - 1ull;
    int nM = 0, nC = 0;
    for (int b = 0; b < BB; ++b) {
        int i = TL[b], j = TL[BB + b];
        const unsigned char* ri = adj + (size_t)i * NN;
        const unsigned char* rj = adj + (size_t)j * NN;
        for (int base = 0; base < NN; base += 32) {
            int u = base + lane;
            bool ai = ri[u] != 0, aj = rj[u] != 0;
            bool ex = ai ^ aj, cm = ai && aj;
            unsigned long long bex = __ballot(ex);
            unsigned long long bcm = __ballot(cm);
            if (ex) {
                int idx = nM + __popcll(bex & lt);
                if (idx < MAXROWS) { rb[idx] = b; ru[idx] = u; rref[idx] = ai ? j : i; }
            }
            if (cm) {
                int idx = nC + __popcll(bcm & lt);
                if (idx < MAXC) { cb[idx] = b; cu[idx] = u; }
            }
            nM += __popcll(bex);
            nC += __popcll(bcm);
        }
    }
    if (lane == 0) {
        counts[0] = nM > MAXROWS ? MAXROWS : nM;
        counts[1] = nC > MAXC ? MAXC : nC;
    }
}

// ---------------- feature build: z[r] = concat(E[u]*E[ref], sum_{v in adj[u]&adj[ref]} E[v]) ----------------
__global__ void build_z(const int* __restrict__ rb, const int* __restrict__ ru,
                        const int* __restrict__ rref, const int* __restrict__ counts,
                        const unsigned int* __restrict__ mask, const float* __restrict__ E,
                        unsigned short* __restrict__ Z) {
    int r = blockIdx.x;
    if (r >= counts[0]) return;
    int u = ru[r], ref = rref[r];
    (void)rb;
    __shared__ unsigned int m[64];
    if (threadIdx.x < 64) m[threadIdx.x] = mask[u * 64 + threadIdx.x] & mask[ref * 64 + threadIdx.x];
    __syncthreads();
    int d = threadIdx.x;
    float p = E[(size_t)u * DD + d] * E[(size_t)ref * DD + d];
    float acc = 0.f;
    for (int wd = 0; wd < 64; ++wd) {
        unsigned int w = m[wd];
        while (w) {
            int bit = __builtin_ctz(w);
            w &= w - 1;
            acc += E[(size_t)(wd * 32 + bit) * DD + d];
        }
    }
    Z[(size_t)r * K1 + d]       = f2bf(p);
    Z[(size_t)r * K1 + DD + d]  = f2bf(acc);
}

// ---------------- WMMA GEMM: C[M,Nout] = relu(A[M,K](bf16) @ Wt[Nout,K]^T + bias) ----------------
__global__ void gemm_bf16_relu(const unsigned short* __restrict__ A,
                               const unsigned short* __restrict__ Bt,
                               const float* __restrict__ bias,
                               float* __restrict__ C,
                               int K, int Nout,
                               const int* __restrict__ cnt, int fixedM) {
    int M = eff_rows(cnt, fixedM);
    int Mp = (M + 15) & ~15;
    int m0 = blockIdx.y * 16;
    if (m0 >= Mp) return;
    int n0 = blockIdx.x * 16;
    int lane = threadIdx.x & 31;
    int hi = lane >> 4, ml = lane & 15;
    const unsigned short* Arow = A  + (size_t)(m0 + ml) * K;
    const unsigned short* Bcol = Bt + (size_t)(n0 + ml) * K;
    v8f acc = {0.f,0.f,0.f,0.f,0.f,0.f,0.f,0.f};
    for (int kb = 0; kb < K; kb += 32) {
        FragBF a, b;
        // A 16x32 bf16 layout: lanes0-15 K=0..7 (v0-3), K=16..23 (v4-7); lanes16-31 +8
        const unsigned short* ap = Arow + kb + 8 * hi;
        #pragma unroll
        for (int p = 0; p < 4; ++p) a.u[p]     = *(const unsigned int*)(ap + 2 * p);
        #pragma unroll
        for (int p = 0; p < 4; ++p) a.u[4 + p] = *(const unsigned int*)(ap + 16 + 2 * p);
        // B 32x16 bf16: lane holds column n; lanes0-15 K=0..7 (v0-3), K=16..23 (v4-7); lanes16-31 +8
        const unsigned short* bp = Bcol + kb + 8 * hi;
        #pragma unroll
        for (int p = 0; p < 4; ++p) b.u[p]     = *(const unsigned int*)(bp + 2 * p);
        #pragma unroll
        for (int p = 0; p < 4; ++p) b.u[4 + p] = *(const unsigned int*)(bp + 16 + 2 * p);
        acc = __builtin_amdgcn_wmma_f32_16x16x32_bf16(false, a.v, false, b.v,
                                                      (short)0, acc, false, false);
    }
    float bi = bias[n0 + ml];
    #pragma unroll
    for (int r = 0; r < 8; ++r) {
        int row = m0 + r + 8 * hi;
        float v = acc[r] + bi;
        C[(size_t)row * Nout + (n0 + ml)] = v > 0.f ? v : 0.f;
    }
}

// ---------------- LayerNorm over 512 (post-relu), write bf16 ----------------
__global__ void ln_bf16(const float* __restrict__ C, const float* __restrict__ g,
                        const float* __restrict__ beta, unsigned short* __restrict__ Hb,
                        const int* __restrict__ cnt, int fixedM) {
    int M = eff_rows(cnt, fixedM);
    int r = blockIdx.x;
    if (r >= M) return;
    __shared__ float red[128];
    int t = threadIdx.x;
    const float* row = C + (size_t)r * HH;
    float v[4]; float s = 0.f;
    #pragma unroll
    for (int q = 0; q < 4; ++q) { v[q] = row[t + 128 * q]; s += v[q]; }
    red[t] = s; __syncthreads();
    for (int st = 64; st > 0; st >>= 1) { if (t < st) red[t] += red[t + st]; __syncthreads(); }
    float mu = red[0] * (1.f / 512.f);
    __syncthreads();
    float s2 = 0.f;
    #pragma unroll
    for (int q = 0; q < 4; ++q) { float d = v[q] - mu; s2 += d * d; }
    red[t] = s2; __syncthreads();
    for (int st = 64; st > 0; st >>= 1) { if (t < st) red[t] += red[t + st]; __syncthreads(); }
    float rstd = rsqrtf(red[0] * (1.f / 512.f) + 1e-5f);
    #pragma unroll
    for (int q = 0; q < 4; ++q) {
        int e = t + 128 * q;
        Hb[(size_t)r * HH + e] = f2bf((v[q] - mu) * rstd * g[e] + beta[e]);
    }
}

// ---------------- f32 -> bf16 elementwise (post-relu hidden) ----------------
__global__ void cvt_bf16(const float* __restrict__ C, unsigned short* __restrict__ Hb,
                         const int* __restrict__ cnt, int fixedM) {
    int M = eff_rows(cnt, fixedM);
    int idx = blockIdx.x * blockDim.x + threadIdx.x;
    if (idx >= M * HH) return;
    Hb[idx] = f2bf(C[idx]);
}

// ---------------- last layer: out[r] = (sigmoid?)(dot(C[r], W4) + b4) ----------------
__global__ void dot_row(const float* __restrict__ C, const float* __restrict__ W4,
                        const float* __restrict__ b4, float* __restrict__ out,
                        int do_sigmoid, const int* __restrict__ cnt, int fixedM) {
    int M = eff_rows(cnt, fixedM);
    int r = blockIdx.x;
    if (r >= M) return;
    int lane = threadIdx.x;
    float s = 0.f;
    for (int k = lane; k < HH; k += 32) s += C[(size_t)r * HH + k] * W4[k];
    for (int o = 16; o > 0; o >>= 1) s += __shfl_xor(s, o, 32);
    if (lane == 0) {
        float l = s + b4[0];
        out[r] = do_sigmoid ? 1.f / (1.f + expf(-l)) : l;
    }
}

// ---------------- allCN[b] = sum_common E[u] + sum_excl w[r]*E[u]  (deterministic scan) ----------------
__global__ void scatter_allcn(const int* __restrict__ rb, const int* __restrict__ ru,
                              const int* __restrict__ cb, const int* __restrict__ cu,
                              const int* __restrict__ counts, const float* __restrict__ w,
                              const float* __restrict__ E, float* __restrict__ allCN) {
    int b = blockIdx.x, d = threadIdx.x;
    int nM = counts[0], nC = counts[1];
    float acc = 0.f;
    for (int r = 0; r < nC; ++r) if (cb[r] == b) acc += E[(size_t)cu[r] * DD + d];
    for (int r = 0; r < nM; ++r) if (rb[r] == b) acc += w[r] * E[(size_t)ru[r] * DD + d];
    allCN[b * DD + d] = acc;
}

// ---------------- X2[b] = concat(E[i]*E[j], allCN[b]) in bf16 ----------------
__global__ void build_x2(const int* __restrict__ TL, const float* __restrict__ E,
                         const float* __restrict__ allCN, unsigned short* __restrict__ X2) {
    int b = blockIdx.x, d = threadIdx.x;
    int i = TL[b], j = TL[BB + b];
    X2[b * K1 + d]      = f2bf(E[(size_t)i * DD + d] * E[(size_t)j * DD + d]);
    X2[b * K1 + DD + d] = f2bf(allCN[b * DD + d]);
}

// ---------------- host side ----------------
extern "C" void kernel_launch(void* const* d_in, const int* in_sizes, int n_in,
                              void* d_out, int out_size, void* d_ws, size_t ws_size,
                              hipStream_t stream) {
    (void)in_sizes; (void)n_in; (void)out_size; (void)ws_size;
    const int*           TL  = (const int*)d_in[0];
    const unsigned char* adj = (const unsigned char*)d_in[1];
    const float*         E   = (const float*)d_in[2];
    const float *nW1=(const float*)d_in[3],  *nb1=(const float*)d_in[4];
    const float *ng =(const float*)d_in[5],  *nbe=(const float*)d_in[6];
    const float *nW2=(const float*)d_in[7],  *nb2=(const float*)d_in[8];
    const float *nW3=(const float*)d_in[9],  *nb3=(const float*)d_in[10];
    const float *nW4=(const float*)d_in[11], *nb4=(const float*)d_in[12];
    const float *mW1=(const float*)d_in[13], *mb1=(const float*)d_in[14];
    const float *mg =(const float*)d_in[15], *mbe=(const float*)d_in[16];
    const float *mW2=(const float*)d_in[17], *mb2=(const float*)d_in[18];
    const float *mW3=(const float*)d_in[19], *mb3=(const float*)d_in[20];
    const float *mW4=(const float*)d_in[21], *mb4=(const float*)d_in[22];
    float* out = (float*)d_out;

    char* ws = (char*)d_ws;
    size_t off = 0;
    auto take = [&](size_t bytes) -> char* {
        char* p = ws + off;
        off += (bytes + 255) & ~(size_t)255;
        return p;
    };
    unsigned int*   mask  = (unsigned int*)  take((size_t)NN * 64 * 4);
    unsigned short* Z     = (unsigned short*)take((size_t)MAXROWS * K1 * 2);
    float*          Cbuf  = (float*)         take((size_t)MAXROWS * HH * 4);
    unsigned short* Hb    = (unsigned short*)take((size_t)MAXROWS * HH * 2);
    float*          wbuf  = (float*)         take((size_t)MAXROWS * 4);
    int* rb   = (int*)take((size_t)MAXROWS * 4);
    int* ru   = (int*)take((size_t)MAXROWS * 4);
    int* rref = (int*)take((size_t)MAXROWS * 4);
    int* cb   = (int*)take((size_t)MAXC * 4);
    int* cu   = (int*)take((size_t)MAXC * 4);
    int* counts = (int*)take(256);
    float* allCN = (float*)take((size_t)BB * DD * 4);
    unsigned short* X2 = (unsigned short*)take((size_t)BB * K1 * 2);
    unsigned short* nW1t = (unsigned short*)take((size_t)HH * K1 * 2);
    unsigned short* nW2t = (unsigned short*)take((size_t)HH * HH * 2);
    unsigned short* nW3t = (unsigned short*)take((size_t)HH * HH * 2);
    unsigned short* mW1t = (unsigned short*)take((size_t)HH * K1 * 2);
    unsigned short* mW2t = (unsigned short*)take((size_t)HH * HH * 2);
    unsigned short* mW3t = (unsigned short*)take((size_t)HH * HH * 2);

    // prep
    zero_u32<<<(MAXROWS * K1 / 2 + 255) / 256, 256, 0, stream>>>((unsigned int*)Z, MAXROWS * K1 / 2);
    pack_adj<<<(NN * 64 + 255) / 256, 256, 0, stream>>>(adj, mask);
    prep_wt<<<(K1 * HH + 255) / 256, 256, 0, stream>>>(nW1, nW1t, K1, HH);
    prep_wt<<<(HH * HH + 255) / 256, 256, 0, stream>>>(nW2, nW2t, HH, HH);
    prep_wt<<<(HH * HH + 255) / 256, 256, 0, stream>>>(nW3, nW3t, HH, HH);
    prep_wt<<<(K1 * HH + 255) / 256, 256, 0, stream>>>(mW1, mW1t, K1, HH);
    prep_wt<<<(HH * HH + 255) / 256, 256, 0, stream>>>(mW2, mW2t, HH, HH);
    prep_wt<<<(HH * HH + 255) / 256, 256, 0, stream>>>(mW3, mW3t, HH, HH);

    // candidates + features
    compact<<<1, 32, 0, stream>>>(TL, adj, rb, ru, rref, cb, cu, counts);
    build_z<<<MAXROWS, 128, 0, stream>>>(rb, ru, rref, counts, mask, E, Z);

    // NCN MLP over exclusive-neighbor rows (sigmoid at the end)
    dim3 gBig(HH / 16, MAXROWS / 16);
    gemm_bf16_relu<<<gBig, 32, 0, stream>>>(Z,  nW1t, nb1, Cbuf, K1, HH, counts, -1);
    ln_bf16<<<MAXROWS, 128, 0, stream>>>(Cbuf, ng, nbe, Hb, counts, -1);
    gemm_bf16_relu<<<gBig, 32, 0, stream>>>(Hb, nW2t, nb2, Cbuf, HH, HH, counts, -1);
    cvt_bf16<<<(MAXROWS * HH + 255) / 256, 256, 0, stream>>>(Cbuf, Hb, counts, -1);
    gemm_bf16_relu<<<gBig, 32, 0, stream>>>(Hb, nW3t, nb3, Cbuf, HH, HH, counts, -1);
    dot_row<<<MAXROWS, 32, 0, stream>>>(Cbuf, nW4, nb4, wbuf, 1, counts, -1);

    // allCN and final features
    scatter_allcn<<<BB, DD, 0, stream>>>(rb, ru, cb, cu, counts, wbuf, E, allCN);
    build_x2<<<BB, DD, 0, stream>>>(TL, E, allCN, X2);

    // final MLP over 16 rows -> d_out
    dim3 gSmall(HH / 16, 1);
    gemm_bf16_relu<<<gSmall, 32, 0, stream>>>(X2, mW1t, mb1, Cbuf, K1, HH, nullptr, BB);
    ln_bf16<<<BB, 128, 0, stream>>>(Cbuf, mg, mbe, Hb, nullptr, BB);
    gemm_bf16_relu<<<gSmall, 32, 0, stream>>>(Hb, mW2t, mb2, Cbuf, HH, HH, nullptr, BB);
    cvt_bf16<<<(BB * HH + 255) / 256, 256, 0, stream>>>(Cbuf, Hb, nullptr, BB);
    gemm_bf16_relu<<<gSmall, 32, 0, stream>>>(Hb, mW3t, mb3, Cbuf, HH, HH, nullptr, BB);
    dot_row<<<BB, 32, 0, stream>>>(Cbuf, mW4, mb4, out, 0, nullptr, BB);
}
